// PixelCNNProg_6201932775596
// MI455X (gfx1250) — matrix operand order repaired and checked by reading, not compiled, not run on hardware
//
#include <hip/hip_runtime.h>
#include <hip/hip_bf16.h>
#include <cstddef>

typedef _Float16 half16 __attribute__((ext_vector_type(16)));
typedef float    f32x8  __attribute__((ext_vector_type(8)));

#define NPOS   2304      // 48*48
#define SLEN   48
#define KDIM   128
#define CIN    160
#define BATCH  8

static __device__ __forceinline__ f32x8 wmma_f16(half16 a, half16 b, f32x8 c) {
    return __builtin_amdgcn_wmma_f32_16x16x32_f16(false, a, false, b, (short)0, c, false, false);
}

// ---------------------------------------------------------------------------
// Kernel 1: fold positional-embedding channels (128..159) + bias into a
// per-projection additive term peadd[proj][kch][n].  Batch-invariant.
// ---------------------------------------------------------------------------
__global__ void pe_kernel(const float* __restrict__ Wq, const float* __restrict__ bq,
                          const float* __restrict__ Wk, const float* __restrict__ bk,
                          const float* __restrict__ Wv, const float* __restrict__ bv,
                          float* __restrict__ peadd)
{
    int idx = blockIdx.x * blockDim.x + threadIdx.x;
    const int total = 3 * KDIM * NPOS;
    if (idx >= total) return;
    int n    = idx % NPOS;
    int kch  = (idx / NPOS) % KDIM;
    int proj = idx / (NPOS * KDIM);
    const float* W    = (proj == 0) ? Wq : (proj == 1) ? Wk : Wv;
    const float* bias = (proj == 0) ? bq : (proj == 1) ? bk : bv;
    int row = n / SLEN, col = n % SLEN;
    float s = bias[kch];
    #pragma unroll 4
    for (int c = 0; c < 32; ++c) {
        int cc  = c & 15;
        int pos = (c < 16) ? row : col;          // px varies over rows, py over cols
        int fi  = cc * SLEN + pos;               // raw reshape [S,16] -> [16,S]
        int p   = fi >> 4, j = fi & 15;
        float val = 0.0f;
        if (p != 0) {
            float ang = (float)p * __powf(10000.0f, -(float)(2 * (j >> 1)) / 16.0f);
            val = (j & 1) ? __cosf(ang) : __sinf(ang);
        }
        s += W[kch * CIN + 128 + c] * val;
    }
    peadd[idx] = s;
}

// ---------------------------------------------------------------------------
// Kernel 2: QKV projection with WMMA.  One wave -> one 16(kch) x 16(n) tile.
// q,k written position-major [n][kch] (for contiguous attention fragments),
// v written channel-major [kch][n].
// ---------------------------------------------------------------------------
__global__ __launch_bounds__(32)
void proj_kernel(const float* __restrict__ x,
                 const float* __restrict__ Wq, const float* __restrict__ Wk,
                 const float* __restrict__ Wv, const float* __restrict__ peadd,
                 _Float16* __restrict__ qT, _Float16* __restrict__ kT,
                 _Float16* __restrict__ vC)
{
    int lane = threadIdx.x;
    int r  = lane & 15;
    int hi = lane >> 4;
    int blk = blockIdx.x;
    int nt = blk % (NPOS / 16); blk /= (NPOS / 16);
    int kt = blk % (KDIM / 16); blk /= (KDIM / 16);
    int b  = blk % BATCH;       blk /= BATCH;
    int proj = blk;                                   // 0=q 1=k 2=v
    const float* W = (proj == 0) ? Wq : (proj == 1) ? Wk : Wv;
    int n0 = nt * 16;

    f32x8 acc = {0.f,0.f,0.f,0.f,0.f,0.f,0.f,0.f};
    for (int kc = 0; kc < 4; ++kc) {                  // 128 channels in 32-chunks
        half16 A, Bm;
        #pragma unroll
        for (int h = 0; h < 16; ++h) {
            int ca = kc * 32 + h + (hi ? 16 : 0);                      // A layout
            A[h] = (_Float16)W[(kt * 16 + r) * CIN + ca];
            int cb = kc * 32 + h + (hi ? 16 : 0);                      // B layout
            Bm[h] = (_Float16)x[((size_t)b * KDIM + cb) * NPOS + n0 + r];
        }
        acc = wmma_f16(A, Bm, acc);
    }
    #pragma unroll
    for (int i = 0; i < 8; ++i) {                     // C layout: row=i+8*hi, col=r
        int kch = kt * 16 + i + hi * 8;
        int n   = n0 + r;
        float v = acc[i] + peadd[((size_t)proj * KDIM + kch) * NPOS + n];
        v = (v > 0.0f) ? v : 0.01f * v;               // leaky relu
        _Float16 hv = (_Float16)v;
        if (proj == 0)      qT[((size_t)b * NPOS + n) * KDIM + kch] = hv;
        else if (proj == 1) kT[((size_t)b * NPOS + n) * KDIM + kch] = hv;
        else                vC[((size_t)b * KDIM + kch) * NPOS + n] = hv;
    }
}

// ---------------------------------------------------------------------------
// Kernel 3: causal flash attention.  One wave per (batch, 16-row query tile).
// S = Q^T K via 8 WMMA per 32-col chunk; online softmax; O += P V via 8 WMMA.
// ---------------------------------------------------------------------------
__global__ __launch_bounds__(32)
void attn_kernel(const _Float16* __restrict__ qT, const _Float16* __restrict__ kT,
                 const _Float16* __restrict__ vC, float* __restrict__ out)
{
    __shared__ _Float16 lp[16 * 32];                  // P tile, C-layout -> A-layout
    int lane = threadIdx.x;
    int r  = lane & 15;
    int hi = lane >> 4;
    int nt = blockIdx.x % (NPOS / 16);
    int b  = blockIdx.x / (NPOS / 16);
    int n0 = nt * 16;
    const _Float16* qb = qT + (size_t)b * NPOS * KDIM;
    const _Float16* kb = kT + (size_t)b * NPOS * KDIM;
    const _Float16* vb = vC + (size_t)b * KDIM * NPOS;

    // Q fragments: A rows = query positions, K-dim = channels.  Register-resident.
    half16 Aq[4];
    #pragma unroll
    for (int kc = 0; kc < 4; ++kc)
        #pragma unroll
        for (int h = 0; h < 16; ++h)
            Aq[kc][h] = qb[(size_t)(n0 + r) * KDIM + kc * 32 + h + (hi ? 16 : 0)];

    f32x8 O[8];
    #pragma unroll
    for (int t = 0; t < 8; ++t) O[t] = (f32x8){0.f,0.f,0.f,0.f,0.f,0.f,0.f,0.f};
    float rmax[8], rsum[8];
    #pragma unroll
    for (int i = 0; i < 8; ++i) { rmax[i] = -1e30f; rsum[i] = 0.0f; }

    const float sc = 0.08838834764831845f;            // 1/sqrt(128)
    int nchunks = (n0 + 16 + 31) / 32;                // causal span, 32-wide chunks

    for (int ch = 0; ch < nchunks; ++ch) {
        int mc = ch * 32;
        f32x8 S0 = (f32x8){0.f,0.f,0.f,0.f,0.f,0.f,0.f,0.f};
        f32x8 S1 = (f32x8){0.f,0.f,0.f,0.f,0.f,0.f,0.f,0.f};
        #pragma unroll
        for (int kc = 0; kc < 4; ++kc) {              // contiguous 32B B-fragments
            half16 B0 = *(const half16*)(kb + (size_t)(mc + r)      * KDIM + kc * 32 + hi * 16);
            half16 B1 = *(const half16*)(kb + (size_t)(mc + 16 + r) * KDIM + kc * 32 + hi * 16);
            S0 = wmma_f16(Aq[kc], B0, S0);
            S1 = wmma_f16(Aq[kc], B1, S1);
        }

        // scale + causal mask + online softmax.  C layout: row = n0+i+8*hi, col = mc(+16)+r
        float nm[8], scl[8];
        #pragma unroll
        for (int i = 0; i < 8; ++i) {
            int n = n0 + i + hi * 8;
            float s0 = S0[i] * sc, s1 = S1[i] * sc;
            if (mc + r      > n) s0 = -1e30f;
            if (mc + 16 + r > n) s1 = -1e30f;
            S0[i] = s0; S1[i] = s1;
            float m = fmaxf(s0, s1);
            m = fmaxf(m, __shfl_xor(m, 1, 32));
            m = fmaxf(m, __shfl_xor(m, 2, 32));
            m = fmaxf(m, __shfl_xor(m, 4, 32));
            m = fmaxf(m, __shfl_xor(m, 8, 32));
            nm[i] = fmaxf(rmax[i], m);
        }
        __syncthreads();                              // protect prior LDS reads (WAR)
        #pragma unroll
        for (int i = 0; i < 8; ++i) {
            float p0 = __expf(S0[i] - nm[i]);
            float p1 = __expf(S1[i] - nm[i]);
            scl[i]   = __expf(rmax[i] - nm[i]);
            rmax[i]  = nm[i];
            float ls = p0 + p1;
            ls += __shfl_xor(ls, 1, 32);
            ls += __shfl_xor(ls, 2, 32);
            ls += __shfl_xor(ls, 4, 32);
            ls += __shfl_xor(ls, 8, 32);
            rsum[i] = rsum[i] * scl[i] + ls;
            lp[(i + hi * 8) * 32 + r]      = (_Float16)p0;
            lp[(i + hi * 8) * 32 + 16 + r] = (_Float16)p1;
        }
        #pragma unroll
        for (int t = 0; t < 8; ++t)
            #pragma unroll
            for (int i = 0; i < 8; ++i)
                O[t][i] *= scl[i];
        __syncthreads();                              // LDS writes -> reads (RAW)

        half16 PA;                                    // P in A layout: row=r, K=m
        #pragma unroll
        for (int h = 0; h < 16; ++h)
            PA[h] = lp[r * 32 + h + (hi ? 16 : 0)];
        #pragma unroll
        for (int t = 0; t < 8; ++t) {                 // 8 v-column tiles of 16
            half16 Bv = *(const half16*)(vb + (size_t)(t * 16 + r) * NPOS + mc + hi * 16);
            O[t] = wmma_f16(PA, Bv, O[t]);
        }
    }

    // epilogue: divide by row sums, store [B][128][48*48]
    #pragma unroll
    for (int t = 0; t < 8; ++t)
        #pragma unroll
        for (int i = 0; i < 8; ++i) {
            int v = t * 16 + r;
            int n = n0 + i + hi * 8;
            out[((size_t)b * KDIM + v) * NPOS + n] = O[t][i] / rsum[i];
        }
}

// ---------------------------------------------------------------------------
extern "C" void kernel_launch(void* const* d_in, const int* in_sizes, int n_in,
                              void* d_out, int out_size, void* d_ws, size_t ws_size,
                              hipStream_t stream) {
    const float* x  = (const float*)d_in[0];
    const float* Wq = (const float*)d_in[1];
    const float* bq = (const float*)d_in[2];
    const float* Wk = (const float*)d_in[3];
    const float* bk = (const float*)d_in[4];
    const float* Wv = (const float*)d_in[5];
    const float* bv = (const float*)d_in[6];
    float* out = (float*)d_out;

    char* ws = (char*)d_ws;
    const size_t PEADD_BYTES = (size_t)3 * KDIM * NPOS * sizeof(float);     // 3.5 MB
    const size_t QKV_BYTES   = (size_t)BATCH * NPOS * KDIM * sizeof(_Float16); // 4.7 MB each
    float*    peadd = (float*)ws;
    _Float16* qT = (_Float16*)(ws + PEADD_BYTES);
    _Float16* kT = (_Float16*)(ws + PEADD_BYTES + QKV_BYTES);
    _Float16* vC = (_Float16*)(ws + PEADD_BYTES + 2 * QKV_BYTES);

    int peN = 3 * KDIM * NPOS;
    pe_kernel<<<(peN + 255) / 256, 256, 0, stream>>>(Wq, bq, Wk, bk, Wv, bv, peadd);
    proj_kernel<<<3 * BATCH * (KDIM / 16) * (NPOS / 16), 32, 0, stream>>>(
        x, Wq, Wk, Wv, peadd, qT, kT, vC);
    attn_kernel<<<BATCH * (NPOS / 16), 32, 0, stream>>>(qT, kT, vC, out);
}
